// RegressionWarmstartClassifier_15522011808341
// MI455X (gfx1250) — matrix operand order repaired and checked by reading, not compiled, not run on hardware
//
#include <hip/hip_runtime.h>

// ---------------- problem constants ----------------
#define NN 50000
#define EE 800000
#define FF 128
#define HH 128
#define CC 10

// ---------------- WMMA types ----------------
typedef __attribute__((ext_vector_type(16))) __bf16 v16bf;
typedef __attribute__((ext_vector_type(8)))  float  v8f;

union Frag {
    v16bf v;
    uint4 q[2];
};

__device__ __forceinline__ unsigned short f2bf(float f) {
    unsigned int u = __float_as_uint(f);
    unsigned int r = u + 0x7FFFu + ((u >> 16) & 1u);   // round-to-nearest-even
    return (unsigned short)(r >> 16);
}

// ---------------- degree / norm ----------------
__global__ void init_deg_kernel(float* deg, int n) {
    int i = blockIdx.x * blockDim.x + threadIdx.x;
    if (i < n) deg[i] = 1.0f;                          // self-loop
}

__global__ void deg_accum_kernel(const long long* __restrict__ ei, float* deg, int nedges) {
    int e = blockIdx.x * blockDim.x + threadIdx.x;
    if (e < nedges) atomicAdd(&deg[(int)ei[nedges + e]], 1.0f);   // dst row
}

__global__ void dinv_kernel(const float* __restrict__ deg, float* dinv, int n) {
    int i = blockIdx.x * blockDim.x + threadIdx.x;
    if (i < n) {
        float d = deg[i];
        dinv[i] = d > 0.0f ? rsqrtf(d) : 0.0f;
    }
}

// ---------------- fp32 -> bf16 convert (4-wide) ----------------
__global__ void cvt_bf16_kernel(const float* __restrict__ x, unsigned short* __restrict__ xb, int n4) {
    int i = blockIdx.x * blockDim.x + threadIdx.x;
    if (i >= n4) return;
    float4 v = ((const float4*)x)[i];
    ushort4 r;
    r.x = f2bf(v.x); r.y = f2bf(v.y); r.z = f2bf(v.z); r.w = f2bf(v.w);
    ((ushort4*)xb)[i] = r;
}

// ---------------- pack 128x128 W into WMMA B-fragment layout ----------------
// tile = kt*8+nt (kt:0..3 over K, nt:0..7 over out cols).
// lane n, element e:  K = kt*32 + (n>=16 ? 16 : 0) + e,  N = nt*16 + (n&15)
// Stored so each lane reads its 16 bf16 (32B) contiguously.
__global__ void pack_w_kernel(const float* __restrict__ W, unsigned short* __restrict__ Wp) {
    int i = blockIdx.x * blockDim.x + threadIdx.x;     // 0..16383
    if (i >= 16384) return;
    int e    = i & 15;
    int lane = (i >> 4) & 31;
    int tile = i >> 9;
    int kt = tile >> 3, nt = tile & 7;
    int K   = kt * 32 + ((lane >= 16) ? 16 : 0) + e;
    int col = nt * 16 + (lane & 15);
    Wp[i] = f2bf(W[K * 128 + col]);
}

// ---------------- bf16 WMMA GEMM: rows x [128,128] -> f32 ----------------
// Each wave computes TWO 16-row strips (32 rows x 128 cols): every LDS B
// fragment feeds two independent WMMAs, and B is double-buffered so the
// ds_load latency overlaps matrix work.
__global__ __launch_bounds__(256) void gemm_bf16_wmma_kernel(
    const unsigned short* __restrict__ A,    // [rows,128] bf16 row-major
    const unsigned short* __restrict__ Wp,   // packed 16384 bf16
    float* __restrict__ out, int nstrips)
{
    __shared__ unsigned short wlds[16384];   // 32 KB staged weights
    const int tid = threadIdx.x;

    // cooperative 32KB load: 2048 uint4 / 256 threads = 8 each
    const uint4* wsrc = (const uint4*)Wp;
    uint4*       wdst = (uint4*)wlds;
#pragma unroll
    for (int i = 0; i < 8; ++i) wdst[tid + 256 * i] = wsrc[tid + 256 * i];
    __syncthreads();

    const int wave = tid >> 5;
    const int lane = tid & 31;
    const int job  = blockIdx.x * 8 + wave;  // one job = 2 strips
    const int s0   = job * 2;
    if (s0 >= nstrips) return;
    const bool two = (s0 + 1) < nstrips;

    const int row0 = s0 * 16;
    const int row1 = two ? row0 + 16 : row0;  // clamp: keep EXEC all-1s for WMMA
    const int m    = lane & 15;
    const int hi   = lane >> 4;               // lane-half
    const unsigned short* arow0 = A + (size_t)(row0 + m) * 128;
    const unsigned short* arow1 = A + (size_t)(row1 + m) * 128;

    v8f acc0[8] = {};
    v8f acc1[8] = {};

#pragma unroll
    for (int kt = 0; kt < 4; ++kt) {
        Frag a0, a1;
        const int c0 = kt * 32 + hi * 16;     // elems 0-15 -> K=c0..c0+15
        a0.q[0] = *(const uint4*)(arow0 + c0);
        a0.q[1] = *(const uint4*)(arow0 + c0 + 8);
        a1.q[0] = *(const uint4*)(arow1 + c0);
        a1.q[1] = *(const uint4*)(arow1 + c0 + 8);

        Frag b[2];
        {   // prologue: fetch nt=0
            const uint4* bp = (const uint4*)(wlds + ((kt * 8 + 0) * 32 + lane) * 16);
            b[0].q[0] = bp[0];
            b[0].q[1] = bp[1];
        }
#pragma unroll
        for (int nt = 0; nt < 8; ++nt) {
            const int cur = nt & 1;
            if (nt < 7) {                     // prefetch next B tile
                const uint4* bp = (const uint4*)(wlds + ((kt * 8 + nt + 1) * 32 + lane) * 16);
                b[cur ^ 1].q[0] = bp[0];
                b[cur ^ 1].q[1] = bp[1];
            }
            acc0[nt] = __builtin_amdgcn_wmma_f32_16x16x32_bf16(
                false, a0.v, false, b[cur].v, (short)0, acc0[nt], false, false);
            acc1[nt] = __builtin_amdgcn_wmma_f32_16x16x32_bf16(
                false, a1.v, false, b[cur].v, (short)0, acc1[nt], false, false);
        }
    }

    // C/D layout: VGPR r, lane l -> M = (l>=16?8:0)+r, N = l&15
#pragma unroll
    for (int nt = 0; nt < 8; ++nt) {
#pragma unroll
        for (int r = 0; r < 8; ++r) {
            out[(size_t)(row0 + hi * 8 + r) * 128 + nt * 16 + m] = acc0[nt][r];
        }
    }
    if (two) {
#pragma unroll
        for (int nt = 0; nt < 8; ++nt) {
#pragma unroll
            for (int r = 0; r < 8; ++r) {
                out[(size_t)(row1 + hi * 8 + r) * 128 + nt * 16 + m] = acc1[nt][r];
            }
        }
    }
}

// ---------------- zero f32 buffer (4-wide) ----------------
__global__ void zero_kernel(float* p, int n4) {
    int i = blockIdx.x * blockDim.x + threadIdx.x;
    if (i < n4) ((float4*)p)[i] = make_float4(0.f, 0.f, 0.f, 0.f);
}

// ---------------- edge aggregation: wave per edge, L2-resident atomics ----------------
__global__ __launch_bounds__(256) void agg_edges_kernel(
    const long long* __restrict__ ei, const float* __restrict__ t,
    const float* __restrict__ dinv, float* __restrict__ a, int nedges)
{
    int wid  = (blockIdx.x * blockDim.x + threadIdx.x) >> 5;
    int lane = threadIdx.x & 31;
    if (wid >= nedges) return;
    int s = (int)ei[wid];
    int d = (int)ei[nedges + wid];
    float nrm = dinv[s] * dinv[d];
    float4 v = *(const float4*)(t + (size_t)s * 128 + lane * 4);
    float* dp = a + (size_t)d * 128 + lane * 4;
    atomicAdd(dp + 0, v.x * nrm);
    atomicAdd(dp + 1, v.y * nrm);
    atomicAdd(dp + 2, v.z * nrm);
    atomicAdd(dp + 3, v.w * nrm);
}

// ---------------- self-loop + bias + ReLU, emit bf16 (feeds next GEMM) ----------------
__global__ void selfbias_relu_bf16_kernel(
    const float* __restrict__ a, const float* __restrict__ t,
    const float* __restrict__ dinv, const float* __restrict__ b,
    unsigned short* __restrict__ hb, int ntot)
{
    int i = blockIdx.x * blockDim.x + threadIdx.x;
    if (i >= ntot) return;
    int node = i >> 7, j = i & 127;
    float di = dinv[node];
    float v = a[i] + t[i] * di * di + b[j];
    hb[i] = f2bf(fmaxf(v, 0.0f));
}

// ---------------- self-loop + bias + ReLU, emit f32 (feeds classifier) ----------------
__global__ void selfbias_relu_f32_kernel(
    const float* __restrict__ a, const float* __restrict__ t,
    const float* __restrict__ dinv, const float* __restrict__ b,
    float* __restrict__ h, int ntot)
{
    int i = blockIdx.x * blockDim.x + threadIdx.x;
    if (i >= ntot) return;
    int node = i >> 7, j = i & 127;
    float di = dinv[node];
    float v = a[i] + t[i] * di * di + b[j];
    h[i] = fmaxf(v, 0.0f);
}

// ---------------- classifier + log_softmax: wave per node ----------------
__global__ __launch_bounds__(256) void classifier_logsoftmax_kernel(
    const float* __restrict__ h, const float* __restrict__ Wc,
    const float* __restrict__ bc, float* __restrict__ out, int n)
{
    int wid  = (blockIdx.x * blockDim.x + threadIdx.x) >> 5;
    int lane = threadIdx.x & 31;
    if (wid >= n) return;
    float4 v = *(const float4*)(h + (size_t)wid * 128 + lane * 4);
    float p[CC];
#pragma unroll
    for (int c = 0; c < CC; ++c) {
        const float* wr = Wc + (lane * 4) * CC + c;
        p[c] = v.x * wr[0] + v.y * wr[CC] + v.z * wr[2 * CC] + v.w * wr[3 * CC];
    }
#pragma unroll
    for (int off = 16; off >= 1; off >>= 1) {
#pragma unroll
        for (int c = 0; c < CC; ++c) p[c] += __shfl_xor(p[c], off, 32);
    }
    if (lane == 0) {
        float m = -1e30f;
#pragma unroll
        for (int c = 0; c < CC; ++c) { p[c] += bc[c]; m = fmaxf(m, p[c]); }
        float s = 0.0f;
#pragma unroll
        for (int c = 0; c < CC; ++c) s += __expf(p[c] - m);
        float lse = m + __logf(s);
#pragma unroll
        for (int c = 0; c < CC; ++c) out[(size_t)wid * CC + c] = p[c] - lse;
    }
}

// ---------------- host-side launch ----------------
static inline size_t alignup(size_t x) { return (x + 255) & ~(size_t)255; }

extern "C" void kernel_launch(void* const* d_in, const int* in_sizes, int n_in,
                              void* d_out, int out_size, void* d_ws, size_t ws_size,
                              hipStream_t stream) {
    const float*     x   = (const float*)d_in[0];
    const long long* ei  = (const long long*)d_in[1];   // int64 [2,E]
    const float*     W1  = (const float*)d_in[2];
    const float*     b1  = (const float*)d_in[3];
    const float*     W2  = (const float*)d_in[4];
    const float*     b2  = (const float*)d_in[5];
    const float*     Wc  = (const float*)d_in[6];
    const float*     bc  = (const float*)d_in[7];
    float*           outp = (float*)d_out;

    // workspace carve-up (buffers aliased across phases)
    char* w = (char*)d_ws;
    float* deg  = (float*)w;            w += alignup((size_t)NN * 4);
    float* dinv = (float*)w;            w += alignup((size_t)NN * 4);
    unsigned short* xb  = (unsigned short*)w;  w += alignup((size_t)NN * FF * 2); // also h1 bf16
    unsigned short* W1p = (unsigned short*)w;  w += alignup((size_t)16384 * 2);
    unsigned short* W2p = (unsigned short*)w;  w += alignup((size_t)16384 * 2);
    float* t  = (float*)w;              w += alignup((size_t)NN * HH * 4);        // gemm out (t0/t1)
    float* a  = (float*)w;              w += alignup((size_t)NN * HH * 4);        // aggregation acc
    float* h2 = (float*)w;              w += alignup((size_t)NN * HH * 4);        // layer-2 activations

    const int ntot = NN * HH;           // 6.4M
    const int n4   = ntot / 4;          // 1.6M
    const int nstrips = NN / 16;        // 3125 exact
    const int njobs   = (nstrips + 1) / 2;

    // degree / normalization
    init_deg_kernel<<<(NN + 255) / 256, 256, 0, stream>>>(deg, NN);
    deg_accum_kernel<<<(EE + 255) / 256, 256, 0, stream>>>(ei, deg, EE);
    dinv_kernel<<<(NN + 255) / 256, 256, 0, stream>>>(deg, dinv, NN);

    // precision staging
    cvt_bf16_kernel<<<(n4 + 255) / 256, 256, 0, stream>>>(x, xb, n4);
    pack_w_kernel<<<64, 256, 0, stream>>>(W1, W1p);
    pack_w_kernel<<<64, 256, 0, stream>>>(W2, W2p);

    // ---- layer 1 ----
    zero_kernel<<<(n4 + 255) / 256, 256, 0, stream>>>(a, n4);
    gemm_bf16_wmma_kernel<<<(njobs + 7) / 8, 256, 0, stream>>>(xb, W1p, t, nstrips);
    agg_edges_kernel<<<(EE + 7) / 8, 256, 0, stream>>>(ei, t, dinv, a, EE);
    selfbias_relu_bf16_kernel<<<(ntot + 255) / 256, 256, 0, stream>>>(a, t, dinv, b1, xb, ntot);

    // ---- layer 2 ----
    zero_kernel<<<(n4 + 255) / 256, 256, 0, stream>>>(a, n4);
    gemm_bf16_wmma_kernel<<<(njobs + 7) / 8, 256, 0, stream>>>(xb, W2p, t, nstrips);
    agg_edges_kernel<<<(EE + 7) / 8, 256, 0, stream>>>(ei, t, dinv, a, EE);
    selfbias_relu_f32_kernel<<<(ntot + 255) / 256, 256, 0, stream>>>(a, t, dinv, b2, h2, ntot);

    // ---- classifier + log_softmax ----
    classifier_logsoftmax_kernel<<<(NN + 7) / 8, 256, 0, stream>>>(h2, Wc, bc, outp, NN);

    (void)in_sizes; (void)n_in; (void)out_size; (void)ws_size;
}
